// CAAttnBlock_10582799418108
// MI455X (gfx1250) — matrix operand order repaired and checked by reading, not compiled, not run on hardware
//
#include <hip/hip_runtime.h>
#include <hip/hip_bf16.h>
#include <math.h>

// ---------------------------------------------------------------------------
// CAAttnBlock for MI455X (gfx1250, wave32).
// GEMM-shaped work (pointwise convs, projections, MLP) runs on
// v_wmma_f32_16x16x32_bf16 with fp32 accumulation. Weight slabs are staged
// into LDS once per block by the Tensor Data Mover (tensor_load_to_lds +
// s_wait_tensorcnt) and broadcast to 8 waves x 4 M-tiles each. A-fragment
// loads for all 4 M-tiles are issued before the WMMA chain so the scheduler
// can stagger load waits instead of draining per-WMMA.
// Irregular work (LN over C, depthwise 3x3, 7x7 local attention with online
// softmax) runs on VALU.
// ---------------------------------------------------------------------------

#define HW   4096
#define CCH  96
#define MTOT 8192   // B * H * W

typedef __bf16 bf16_t;
typedef __attribute__((ext_vector_type(16))) __bf16 v16bf;
typedef __attribute__((ext_vector_type(8)))  __bf16 v8bf;
typedef __attribute__((ext_vector_type(8)))  float  v8f;
typedef __attribute__((ext_vector_type(4)))  unsigned int u32x4;
typedef __attribute__((ext_vector_type(8)))  int i32x8;
typedef __attribute__((ext_vector_type(4)))  int i32x4;

// ---- input indices (jax tree flatten: sorted dict keys, recursively) ------
enum {
  IN_XA = 0, IN_XB = 1,
  P_KA_DWB, P_KA_DWW, P_KA_PWB, P_KA_PWW,
  P_KB_DWB, P_KB_DWW, P_KB_PWB, P_KB_PWW,
  P_MLPA_FC1B, P_MLPA_FC1W, P_MLPA_FC2B, P_MLPA_FC2W,
  P_MLPB_FC1B, P_MLPB_FC1W, P_MLPB_FC2B, P_MLPB_FC2W,
  P_NA1_B, P_NA1_G, P_NA2_B, P_NA2_G,
  P_NB1_B, P_NB1_G, P_NB2_B, P_NB2_G,
  P_PROJA_B, P_PROJA_W, P_PROJB_B, P_PROJB_W,
  P_QA_DWB, P_QA_DWW, P_QA_PWB, P_QA_PWW,
  P_QB_DWB, P_QB_DWW, P_QB_PWB, P_QB_PWW,
  P_QC_DWB, P_QC_DWW, P_QC_PWB, P_QC_PWW,
  P_VA_DWB, P_VA_DWW, P_VA_PWB, P_VA_PWW,
  P_VB_DWB, P_VB_DWW, P_VB_PWB, P_VB_PWW
};

// ---------------------------------------------------------------------------
// fp32 -> bf16 weight conversion
__global__ void k_cvt(const float* __restrict__ s, bf16_t* __restrict__ d, int n) {
  int i = blockIdx.x * blockDim.x + threadIdx.x;
  if (i < n) d[i] = (bf16_t)s[i];
}

// pack proj weight (96 x 192) -> effective (96 x 64): cols [0:32] and [96:128]
__global__ void k_pack_proj(const float* __restrict__ w, bf16_t* __restrict__ d) {
  int i = blockIdx.x * blockDim.x + threadIdx.x;
  if (i >= 96 * 64) return;
  int o = i >> 6, c = i & 63;
  float v = (c < 32) ? w[o * 192 + c] : w[o * 192 + 96 + (c - 32)];
  d[i] = (bf16_t)v;
}

// ---------------------------------------------------------------------------
// LayerNorm(channel) on both streams + |na-nb|, planar (B,C,HW) fp32 layout.
__global__ void k_ln1(const float* __restrict__ xa, const float* __restrict__ xb,
                      const float* __restrict__ ga, const float* __restrict__ ba,
                      const float* __restrict__ gb, const float* __restrict__ bb,
                      float* __restrict__ na, float* __restrict__ nb,
                      float* __restrict__ dd) {
  int t = blockIdx.x * blockDim.x + threadIdx.x;
  if (t >= MTOT) return;
  int b = t >> 12, pix = t & (HW - 1);
  size_t base = (size_t)b * CCH * HW + pix;
  float sa = 0.f, qa = 0.f, sb = 0.f, qb = 0.f;
  for (int c = 0; c < CCH; c++) {
    float va = xa[base + (size_t)c * HW];
    float vb = xb[base + (size_t)c * HW];
    sa += va; qa += va * va; sb += vb; qb += vb * vb;
  }
  float ma = sa / CCH, mb = sb / CCH;
  float ra = rsqrtf(qa / CCH - ma * ma + 1e-5f);
  float rb = rsqrtf(qb / CCH - mb * mb + 1e-5f);
  for (int c = 0; c < CCH; c++) {
    size_t idx = base + (size_t)c * HW;
    float va = (xa[idx] - ma) * ra * ga[c] + ba[c];
    float vb = (xb[idx] - mb) * rb * gb[c] + bb[c];
    na[idx] = va; nb[idx] = vb; dd[idx] = fabsf(va - vb);
  }
}

// LayerNorm, single stream, bf16 token-major (M x 96) output for GEMM A.
__global__ void k_ln2(const float* __restrict__ x, const float* __restrict__ g,
                      const float* __restrict__ b, bf16_t* __restrict__ y) {
  int t = blockIdx.x * blockDim.x + threadIdx.x;
  if (t >= MTOT) return;
  int bz = t >> 12, pix = t & (HW - 1);
  size_t base = (size_t)bz * CCH * HW + pix;
  float s = 0.f, q = 0.f;
  for (int c = 0; c < CCH; c++) { float v = x[base + (size_t)c * HW]; s += v; q += v * v; }
  float m = s / CCH;
  float r = rsqrtf(q / CCH - m * m + 1e-5f);
  for (int c = 0; c < CCH; c++) {
    float v = (x[base + (size_t)c * HW] - m) * r * g[c] + b[c];
    y[(size_t)t * CCH + c] = (bf16_t)v;
  }
}

// ---------------------------------------------------------------------------
// depthwise 3x3 + bias; planar fp32 in -> bf16 token-major (M x 96) out
__global__ void k_dw(const float* __restrict__ src, const float* __restrict__ w,
                     const float* __restrict__ bias, bf16_t* __restrict__ out) {
  int i = blockIdx.x * blockDim.x + threadIdx.x;   // (b*96+c)*4096 + pix
  if (i >= 2 * CCH * HW) return;
  int pix = i & (HW - 1);
  int c = (i >> 12) % CCH;
  int b = i / (CCH * HW);
  int py = pix >> 6, px = pix & 63;
  const float* sp = src + ((size_t)b * CCH + c) * HW;
  const float* wp = w + c * 9;
  float acc = bias[c];
  for (int dy = -1; dy <= 1; dy++) {
    int yy = py + dy; if ((unsigned)yy >= 64u) continue;
    for (int dx = -1; dx <= 1; dx++) {
      int xx = px + dx; if ((unsigned)xx >= 64u) continue;
      acc += wp[(dy + 1) * 3 + (dx + 1)] * sp[yy * 64 + xx];
    }
  }
  int m = b * HW + pix;
  out[(size_t)m * CCH + c] = (bf16_t)acc;
}

// ---------------------------------------------------------------------------
// WMMA GEMM: Y(MxN) = A(MxK,bf16 row-major) * W(NxK,bf16 row-major)^T
// Block = 256 threads = 8 waves; block owns one 16-wide N-tile and a 512-row
// M-superblock. Wave 0 DMAs the 16xK weight slab into LDS with the Tensor
// Data Mover (TENSORcnt), then each wave computes 4 M-tiles (64x16 output)
// with a shared LDS B fragment per K-step -> 4 independent WMMAs per step.
// Fragment layouts used here (A and B share a contiguous K order):
//   A: lane l<16 row=tm+l, 16 contiguous Ks starting at k0 + h*16 with h = lane>>4
//   B: lane col=tn+(l&15), 16 contiguous Ks starting at k0 + h*16
//   C/D: lane col n=tn+(l&15); vgpr r -> row m = tm + r + 8*h
// EPI: 0 = fp32 token-major (+bias)         (pointwise conv, N=32)
//      1 = fp32 planar (B,C,HW) +bias+resid (projection / fc2)
//      2 = tanh-GELU -> bf16 token-major    (fc1)
template <int EPI>
__global__ void k_gemm(const bf16_t* __restrict__ A, const bf16_t* __restrict__ W,
                       const float* __restrict__ bias, int M, int N, int K,
                       float* __restrict__ outF, bf16_t* __restrict__ outB,
                       const float* __restrict__ resid) {
  extern __shared__ __align__(32) char smem_raw[];
  bf16_t* sW = (bf16_t*)smem_raw;                 // 16 x K bf16, offset 0 in LDS

  int wave = threadIdx.x >> 5;
  int lane = threadIdx.x & 31;
  int mgroups = M >> 9;                            // 512 rows per block
  int tn = (blockIdx.x / mgroups) << 4;
  int mbase = (blockIdx.x % mgroups) << 9;

  if (wave == 0) {
    // --- TDM: load W rows [tn, tn+16) x K bf16 into LDS offset 0 -----------
    // D# group0/group1 per CDNA5 ISA sec 8: 2D tile, data_size=2B.
#if defined(__HIP_DEVICE_COMPILE__)
    unsigned long long ga = (unsigned long long)(const void*)(W + (size_t)tn * K);
    unsigned td0 = (unsigned)K, td1 = 16u;
    unsigned tile0 = (unsigned)K, tile1 = 16u;
    unsigned long long st0 = (unsigned long long)K;
    u32x4 g0;
    g0[0] = 1u;                                    // count=1, user mode
    g0[1] = 0u;                                    // lds_addr = 0
    g0[2] = (unsigned)ga;                          // global_addr[31:0]
    g0[3] = (unsigned)(ga >> 32) | (2u << 30);     // global_addr[56:32] | type=2
    i32x8 g1;
    g1[0] = (int)(1u << 16);                       // wg_mask=0, data_size=1 (2B)
    g1[1] = (int)((td0 & 0xFFFFu) << 16);          // tensor_dim0[15:0]
    g1[2] = (int)((td0 >> 16) | ((td1 & 0xFFFFu) << 16)); // dim0 hi | dim1 lo
    g1[3] = (int)((td1 >> 16) | (tile0 << 16));    // dim1 hi | tile_dim0
    g1[4] = (int)tile1;                            // tile_dim1 | tile_dim2=0
    g1[5] = (int)(unsigned)st0;                    // tensor_dim0_stride[31:0]
    g1[6] = (int)((unsigned)(st0 >> 32) & 0xFFFFu);// stride0 hi | stride1 lo=0
    g1[7] = 0;
    i32x4 z4 = {0, 0, 0, 0};
    i32x8 z8 = {0, 0, 0, 0, 0, 0, 0, 0};
    // 6-arg toolchain variant: (g0, g1, g2, g3, g4, cpol)
    __builtin_amdgcn_tensor_load_to_lds(g0, g1, z4, z4, z8, 0);
    __builtin_amdgcn_s_wait_tensorcnt(0);
#endif
  }
  __syncthreads();

  int half = lane >> 4, l15 = lane & 15;
  const bf16_t* sWrow = sW + (size_t)l15 * K + half * 16;
  int tm = mbase + wave * 64;
  const bf16_t* ar[4];
#pragma unroll
  for (int t = 0; t < 4; t++)
    ar[t] = A + (size_t)(tm + t * 16 + l15) * K + half * 16;

  v8f acc[4] = {};
  for (int k0 = 0; k0 < K; k0 += 32) {
    // Issue the shared B fragment (LDS) and all 4 A fragments (global) first
    // so the scheduler can stagger s_wait_loadcnt across the 4 WMMAs.
    v16bf bv = *(const v16bf*)(sWrow + k0);
    union { v16bf v; v8bf h[2]; } ua[4];
#pragma unroll
    for (int t = 0; t < 4; t++) {
      ua[t].h[0] = *(const v8bf*)(ar[t] + k0);     // K = k0 + h*16 + 0..7
      ua[t].h[1] = *(const v8bf*)(ar[t] + k0 + 8); // K = k0 + h*16 + 8..15
    }
#pragma unroll
    for (int t = 0; t < 4; t++)
      acc[t] = __builtin_amdgcn_wmma_f32_16x16x32_bf16(
          false, ua[t].v, false, bv, (short)0, acc[t], false, false);
  }

  int n = tn + l15;
  float bn = bias[n];
#pragma unroll
  for (int t = 0; t < 4; t++) {
#pragma unroll
    for (int r = 0; r < 8; r++) {
      int m = tm + t * 16 + r + (half << 3);
      float v = acc[t][r] + bn;
      if (EPI == 0) {
        outF[(size_t)m * N + n] = v;
      } else if (EPI == 1) {
        size_t p = ((size_t)(m >> 12) * CCH + n) * HW + (m & (HW - 1));
        outF[p] = v + resid[p];
      } else {
        float g = 0.5f * v * (1.0f + tanhf(0.7978845608028654f * (v + 0.044715f * v * v * v)));
        outB[(size_t)m * N + n] = (bf16_t)g;
      }
    }
  }
}

// ---------------------------------------------------------------------------
// 7x7 local attention, online softmax. One thread = (token, head).
// heads 0-1: self-a; 2-3: self-b; 4-5: cross (keys = [ka;kb], values = |va-vb|).
// Zero-padded neighbors contribute score 0 (not -inf) with zero value.
__global__ void k_attn(const float* __restrict__ qa, const float* __restrict__ ka,
                       const float* __restrict__ va, const float* __restrict__ qb,
                       const float* __restrict__ kb, const float* __restrict__ vb,
                       const float* __restrict__ qc,
                       bf16_t* __restrict__ cat_a, bf16_t* __restrict__ cat_b) {
  int tid = blockIdx.x * blockDim.x + threadIdx.x;
  if (tid >= 6 * MTOT) return;
  int head = tid >> 13;
  int m = tid & (MTOT - 1);
  int b = m >> 12, pix = m & (HW - 1);
  int py = pix >> 6, px = pix & 63;
  int strm = head >> 1;
  int hoff = (head & 1) << 4;

  const float* Q = (strm == 0) ? qa : (strm == 1) ? qb : qc;
  float q[16];
  const float* qp = Q + (size_t)m * 32 + hoff;
#pragma unroll
  for (int j = 0; j < 16; j++) q[j] = qp[j] * 0.25f;   // 1/sqrt(16)

  float mx = -1e30f, l = 0.f, acc[16];
#pragma unroll
  for (int j = 0; j < 16; j++) acc[j] = 0.f;

  int nseg = (strm == 2) ? 2 : 1;
  for (int seg = 0; seg < nseg; seg++) {
    const float* Kt = (strm == 0) ? ka : (strm == 1) ? kb : (seg == 0 ? ka : kb);
    for (int dy = -3; dy <= 3; dy++) {
      int yy = py + dy;
      bool rowok = (unsigned)yy < 64u;
      for (int dx = -3; dx <= 3; dx++) {
        int xx = px + dx;
        bool ok = rowok && ((unsigned)xx < 64u);
        float s = 0.f;
        float vv[16];
        if (ok) {
          int nm = b * HW + yy * 64 + xx;
          const float* kp = Kt + (size_t)nm * 32 + hoff;
#pragma unroll
          for (int j = 0; j < 16; j++) s += q[j] * kp[j];
          if (strm == 0) {
            const float* vp = va + (size_t)nm * 32 + hoff;
#pragma unroll
            for (int j = 0; j < 16; j++) vv[j] = vp[j];
          } else if (strm == 1) {
            const float* vp = vb + (size_t)nm * 32 + hoff;
#pragma unroll
            for (int j = 0; j < 16; j++) vv[j] = vp[j];
          } else {
            const float* pva = va + (size_t)nm * 32 + hoff;
            const float* pvb = vb + (size_t)nm * 32 + hoff;
#pragma unroll
            for (int j = 0; j < 16; j++) vv[j] = fabsf(pva[j] - pvb[j]);
          }
        } else {
#pragma unroll
          for (int j = 0; j < 16; j++) vv[j] = 0.f;
        }
        float nmx = fmaxf(mx, s);
        float cf = __expf(mx - nmx);
        float p = __expf(s - nmx);
        l = l * cf + p;
#pragma unroll
        for (int j = 0; j < 16; j++) acc[j] = acc[j] * cf + p * vv[j];
        mx = nmx;
      }
    }
  }
  float inv = 1.f / l;
  if (strm == 0) {
    bf16_t* o = cat_a + (size_t)m * 64 + hoff;
#pragma unroll
    for (int j = 0; j < 16; j++) o[j] = (bf16_t)(acc[j] * inv);
  } else if (strm == 1) {
    bf16_t* o = cat_b + (size_t)m * 64 + hoff;
#pragma unroll
    for (int j = 0; j < 16; j++) o[j] = (bf16_t)(acc[j] * inv);
  } else {
    bf16_t* o1 = cat_a + (size_t)m * 64 + 32 + hoff;
    bf16_t* o2 = cat_b + (size_t)m * 64 + 32 + hoff;
#pragma unroll
    for (int j = 0; j < 16; j++) {
      bf16_t z = (bf16_t)(acc[j] * inv);
      o1[j] = z; o2[j] = z;
    }
  }
}

// ---------------------------------------------------------------------------
extern "C" void kernel_launch(void* const* d_in, const int* in_sizes, int n_in,
                              void* d_out, int out_size, void* d_ws, size_t ws_size,
                              hipStream_t stream) {
  (void)in_sizes; (void)n_in; (void)out_size; (void)ws_size;
  const float* xa = (const float*)d_in[IN_XA];
  const float* xb = (const float*)d_in[IN_XB];
  auto P = [&](int i) { return (const float*)d_in[i]; };

  char* ws = (char*)d_ws;
  size_t off = 0;
  auto alloc = [&](size_t bytes) -> void* {
    void* p = ws + off;
    off += (bytes + 255) & ~(size_t)255;
    return p;
  };

  const size_t PLANE = (size_t)2 * CCH * HW;           // 786432 elems
  float*  na    = (float*)alloc(PLANE * 4);
  float*  nb    = (float*)alloc(PLANE * 4);
  float*  dd    = (float*)alloc(PLANE * 4);
  bf16_t* dwout = (bf16_t*)alloc((size_t)7 * MTOT * CCH * 2);
  float*  pwout = (float*)alloc((size_t)7 * MTOT * 32 * 4);
  bf16_t* cat_a = (bf16_t*)alloc((size_t)MTOT * 64 * 2);
  bf16_t* cat_b = (bf16_t*)alloc((size_t)MTOT * 64 * 2);
  float*  x1a   = (float*)alloc(PLANE * 4);
  float*  x1b   = (float*)alloc(PLANE * 4);
  bf16_t* y2a   = (bf16_t*)alloc((size_t)MTOT * CCH * 2);
  bf16_t* y2b   = (bf16_t*)alloc((size_t)MTOT * CCH * 2);
  bf16_t* ha    = (bf16_t*)alloc((size_t)MTOT * 384 * 2);
  bf16_t* hb    = (bf16_t*)alloc((size_t)MTOT * 384 * 2);
  bf16_t* pwW   = (bf16_t*)alloc((size_t)7 * 32 * CCH * 2);
  bf16_t* prWa  = (bf16_t*)alloc((size_t)CCH * 64 * 2);
  bf16_t* prWb  = (bf16_t*)alloc((size_t)CCH * 64 * 2);
  bf16_t* f1Wa  = (bf16_t*)alloc((size_t)384 * CCH * 2);
  bf16_t* f1Wb  = (bf16_t*)alloc((size_t)384 * CCH * 2);
  bf16_t* f2Wa  = (bf16_t*)alloc((size_t)CCH * 384 * 2);
  bf16_t* f2Wb  = (bf16_t*)alloc((size_t)CCH * 384 * 2);

  // ---- weight prep (bf16). Pointwise: only output channels 0..31 are ever
  // consumed downstream -> convert just the first 32 rows (contiguous 3072).
  const int pw_w_idx[7] = {P_QA_PWW, P_KA_PWW, P_VA_PWW, P_QB_PWW, P_KB_PWW, P_VB_PWW, P_QC_PWW};
  const int pw_b_idx[7] = {P_QA_PWB, P_KA_PWB, P_VA_PWB, P_QB_PWB, P_KB_PWB, P_VB_PWB, P_QC_PWB};
  const int dw_w_idx[7] = {P_QA_DWW, P_KA_DWW, P_VA_DWW, P_QB_DWW, P_KB_DWW, P_VB_DWW, P_QC_DWW};
  const int dw_b_idx[7] = {P_QA_DWB, P_KA_DWB, P_VA_DWB, P_QB_DWB, P_KB_DWB, P_VB_DWB, P_QC_DWB};
  for (int t = 0; t < 7; t++)
    k_cvt<<<12, 256, 0, stream>>>(P(pw_w_idx[t]), pwW + (size_t)t * 32 * CCH, 32 * CCH);
  k_cvt<<<144, 256, 0, stream>>>(P(P_MLPA_FC1W), f1Wa, 384 * CCH);
  k_cvt<<<144, 256, 0, stream>>>(P(P_MLPB_FC1W), f1Wb, 384 * CCH);
  k_cvt<<<144, 256, 0, stream>>>(P(P_MLPA_FC2W), f2Wa, CCH * 384);
  k_cvt<<<144, 256, 0, stream>>>(P(P_MLPB_FC2W), f2Wb, CCH * 384);
  k_pack_proj<<<24, 256, 0, stream>>>(P(P_PROJA_W), prWa);
  k_pack_proj<<<24, 256, 0, stream>>>(P(P_PROJB_W), prWb);

  // ---- LN1 (+|na-nb|)
  k_ln1<<<32, 256, 0, stream>>>(xa, xb, P(P_NA1_G), P(P_NA1_B), P(P_NB1_G), P(P_NB1_B),
                                na, nb, dd);

  // ---- depthwise 3x3 (7 branches) -> bf16 token-major GEMM A
  const float* dwsrc[7] = {na, na, na, nb, nb, nb, dd};
  for (int t = 0; t < 7; t++)
    k_dw<<<3072, 256, 0, stream>>>(dwsrc[t], P(dw_w_idx[t]), P(dw_b_idx[t]),
                                   dwout + (size_t)t * MTOT * CCH);

  // ---- pointwise 1x1 as WMMA GEMM (M=8192, N=32, K=96)
  // grid = tilesN * (M/512); dynamic LDS = 16*K*2 bytes (TDM weight slab)
  for (int t = 0; t < 7; t++)
    k_gemm<0><<<2 * 16, 256, 16 * CCH * 2, stream>>>(
        dwout + (size_t)t * MTOT * CCH, pwW + (size_t)t * 32 * CCH,
        P(pw_b_idx[t]), MTOT, 32, CCH,
        pwout + (size_t)t * MTOT * 32, nullptr, nullptr);

  // ---- local window attention (self-a, self-b, cross)
  const float* pq = pwout;
  const float* pk = pwout + (size_t)1 * MTOT * 32;
  const float* pv = pwout + (size_t)2 * MTOT * 32;
  const float* qB = pwout + (size_t)3 * MTOT * 32;
  const float* kB = pwout + (size_t)4 * MTOT * 32;
  const float* vB = pwout + (size_t)5 * MTOT * 32;
  const float* qC = pwout + (size_t)6 * MTOT * 32;
  k_attn<<<192, 256, 0, stream>>>(pq, pk, pv, qB, kB, vB, qC, cat_a, cat_b);

  // ---- projection GEMMs (M=8192, N=96, K=64) + residual -> planar x1
  k_gemm<1><<<6 * 16, 256, 16 * 64 * 2, stream>>>(cat_a, prWa, P(P_PROJA_B),
                                                  MTOT, CCH, 64, x1a, nullptr, xa);
  k_gemm<1><<<6 * 16, 256, 16 * 64 * 2, stream>>>(cat_b, prWb, P(P_PROJB_B),
                                                  MTOT, CCH, 64, x1b, nullptr, xb);

  // ---- LN2 -> bf16 token-major
  k_ln2<<<32, 256, 0, stream>>>(x1a, P(P_NA2_G), P(P_NA2_B), y2a);
  k_ln2<<<32, 256, 0, stream>>>(x1b, P(P_NB2_G), P(P_NB2_B), y2b);

  // ---- MLP: fc1 (96->384, GELU) then fc2 (384->96, +bias+residual -> d_out)
  k_gemm<2><<<24 * 16, 256, 16 * CCH * 2, stream>>>(y2a, f1Wa, P(P_MLPA_FC1B),
                                                    MTOT, 384, CCH, nullptr, ha, nullptr);
  k_gemm<2><<<24 * 16, 256, 16 * CCH * 2, stream>>>(y2b, f1Wb, P(P_MLPB_FC1B),
                                                    MTOT, 384, CCH, nullptr, hb, nullptr);
  float* out_a = (float*)d_out;
  float* out_b = out_a + PLANE;
  k_gemm<1><<<6 * 16, 256, 16 * 384 * 2, stream>>>(ha, f2Wa, P(P_MLPA_FC2B),
                                                   MTOT, CCH, 384, out_a, nullptr, x1a);
  k_gemm<1><<<6 * 16, 256, 16 * 384 * 2, stream>>>(hb, f2Wb, P(P_MLPB_FC2B),
                                                   MTOT, CCH, 384, out_b, nullptr, x1b);
}
